// GptOssAttention_86449101734357
// MI455X (gfx1250) — matrix operand order repaired and checked by reading, not compiled, not run on hardware
//
#include <hip/hip_runtime.h>
#include <hip/hip_bf16.h>

typedef __attribute__((ext_vector_type(16))) __bf16 v16bf;
typedef __attribute__((ext_vector_type(8)))  __bf16 v8bf;
typedef __attribute__((ext_vector_type(8)))  float  v8f;

#define S_LEN  4096
#define HID_C  2880
#define HQ_C   64
#define HKV_C  8
#define D_HEAD 64
#define W_WIN  128
#define NB_C   (S_LEN / W_WIN)        // 32
#define QN_C   (HQ_C * D_HEAD)        // 4096
#define KVN_C  (HKV_C * D_HEAD)       // 512
#define SCALE_C 0.125f                // 64^-0.5
#define NEGINF  -1e30f

__device__ __forceinline__ unsigned short f2bf(float f) {
  union { float f; unsigned u; } a; a.f = f;
  unsigned u = a.u;
  u += 0x7FFFu + ((u >> 16) & 1u);   // round-to-nearest-even
  return (unsigned short)(u >> 16);
}

// A-fragment (16x32 bf16): lane l -> row (l&15); K = base(l<16?0:8) + {0..7} and {16..23}
__device__ __forceinline__ v16bf load_fragA(const unsigned short* p0, int ld) {
  int lane = threadIdx.x & 31;
  const unsigned short* p = p0 + (lane & 15) * ld + ((lane < 16) ? 0 : 8);
  union { v16bf v; v8bf h[2]; } u;
  u.h[0] = *(const v8bf*)(p);
  u.h[1] = *(const v8bf*)(p + 16);
  return u.v;
}

// B-fragment loaded from Bt (N x K row-major): lane l -> column n=(l&15);
// K = base(l<16?0:8) + {0..7} and {16..23}
__device__ __forceinline__ v16bf load_fragB(const unsigned short* p0, int ld) {
  int lane = threadIdx.x & 31;
  const unsigned short* p = p0 + (lane & 15) * ld + ((lane < 16) ? 0 : 8);
  union { v16bf v; v8bf h[2]; } u;
  u.h[0] = *(const v8bf*)(p);
  u.h[1] = *(const v8bf*)(p + 16);
  return u.v;
}

__device__ __forceinline__ v8f wmma_bf16(v16bf a, v16bf b, v8f c) {
  return __builtin_amdgcn_wmma_f32_16x16x32_bf16(false, a, false, b, (short)0, c, false, false);
}

// ---------------------------------------------------------------------------
// elementwise conversions
// ---------------------------------------------------------------------------
__global__ void cvt_bf16_kernel(const float* __restrict__ in,
                                unsigned short* __restrict__ out, int n) {
  int i = blockIdx.x * 256 + threadIdx.x;
  if (i < n) out[i] = f2bf(in[i]);
}

// in: [Kd][Nd] f32  ->  out: [Nd][Kd] bf16
__global__ void cvt_transpose_kernel(const float* __restrict__ in,
                                     unsigned short* __restrict__ out,
                                     int Kd, int Nd) {
  int i = blockIdx.x * 256 + threadIdx.x;
  if (i < Kd * Nd) {
    int nidx = i / Kd;
    int kidx = i - nidx * Kd;
    out[i] = f2bf(in[kidx * Nd + nidx]);
  }
}

// ---------------------------------------------------------------------------
// GEMM core: acc[2][4] (32x64 per wave) over K, bf16 WMMA, f32 accumulate.
// Block tile 256x64 (8 waves x 32 rows). Bt tile double-buffered in LDS:
// one barrier per K step; next tile's global->LDS copy overlaps WMMAs.
// ---------------------------------------------------------------------------
__device__ __forceinline__ void gemm_core(const unsigned short* __restrict__ A,
                                          const unsigned short* __restrict__ Bt,
                                          int K, int m0, int n0,
                                          unsigned short* sB,      // [2][64*32]
                                          v8f acc[2][4]) {
  int t = threadIdx.x;
  int brow = t >> 2;
  int bcol = (t & 3) * 8;
  const unsigned short* bSrc = Bt + (n0 + brow) * K + bcol;
  const unsigned short* a0p = A + m0 * K;
  const unsigned short* a1p = A + (m0 + 16) * K;

#pragma unroll
  for (int mt = 0; mt < 2; ++mt)
#pragma unroll
    for (int nt = 0; nt < 4; ++nt) acc[mt][nt] = (v8f)0.0f;

  // preload stage 0
  *(v8bf*)(&sB[t * 8]) = *(const v8bf*)(bSrc);
  int cur = 0;
  for (int kk = 0; kk < K; kk += 32) {
    __syncthreads();                     // publishes sB[cur]; guards sB[cur^1] reuse
    if (kk + 32 < K)
      *(v8bf*)(&sB[(cur ^ 1) * 2048 + t * 8]) = *(const v8bf*)(bSrc + kk + 32);
    v16bf a0 = load_fragA(a0p + kk, K);
    v16bf a1 = load_fragA(a1p + kk, K);
#pragma unroll
    for (int nt = 0; nt < 4; ++nt) {
      v16bf b = load_fragB(&sB[cur * 2048 + nt * 16 * 32], 32);
      acc[0][nt] = wmma_bf16(a0, b, acc[0][nt]);
      acc[1][nt] = wmma_bf16(a1, b, acc[1][nt]);
    }
    cur ^= 1;
  }
}

// C[M][N] f32 = A*Bt^T + bias   (used for the O-projection)
__global__ __launch_bounds__(256)
void gemm_bias_f32_kernel(const unsigned short* __restrict__ A,
                          const unsigned short* __restrict__ Bt,
                          const float* __restrict__ bias,
                          float* __restrict__ C, int N, int K) {
  __shared__ unsigned short sB[2 * 2048];
  int wave = threadIdx.x >> 5;
  int lane = threadIdx.x & 31;
  int m0 = blockIdx.x * 256 + wave * 32;
  int n0 = blockIdx.y * 64;
  v8f acc[2][4];
  gemm_core(A, Bt, K, m0, n0, sB, acc);
  int col  = lane & 15;
  int rofs = (lane < 16) ? 0 : 8;
#pragma unroll
  for (int mt = 0; mt < 2; ++mt)
#pragma unroll
    for (int nt = 0; nt < 4; ++nt) {
      float bval = bias[n0 + nt * 16 + col];
#pragma unroll
      for (int r = 0; r < 8; ++r)
        C[(m0 + mt * 16 + r + rofs) * N + n0 + nt * 16 + col] = acc[mt][nt][r] + bval;
    }
}

// Fused QKV-projection + bias + RoPE epilogue -> bf16 [H][S][64].
// One 64-col block == one head; RoPE pair (d, d+32) == acc tiles (nt, nt^2)
// in the same lane/row, so rotation is pure register math.
__global__ __launch_bounds__(256)
void gemm_rope_bf16_kernel(const unsigned short* __restrict__ A,
                           const unsigned short* __restrict__ Bt,
                           const float* __restrict__ bias,
                           const float* __restrict__ fc,   // [S][64] = cos(32)|sin(32)
                           unsigned short* __restrict__ outp,
                           int K, float scale) {
  __shared__ unsigned short sB[2 * 2048];
  int wave = threadIdx.x >> 5;
  int lane = threadIdx.x & 31;
  int m0 = blockIdx.x * 256 + wave * 32;
  int n0 = blockIdx.y * 64;
  v8f acc[2][4];
  gemm_core(A, Bt, K, m0, n0, sB, acc);
  int col  = lane & 15;
  int rofs = (lane < 16) ? 0 : 8;
  int h = n0 >> 6;
  float b0 = bias[n0 + col],      b1 = bias[n0 + 16 + col];
  float b2 = bias[n0 + 32 + col], b3 = bias[n0 + 48 + col];
#pragma unroll
  for (int mt = 0; mt < 2; ++mt)
#pragma unroll
    for (int r = 0; r < 8; ++r) {
      int s = m0 + mt * 16 + r + rofs;
      float c0  = fc[s * 64 + col],      c1  = fc[s * 64 + 16 + col];
      float sn0 = fc[s * 64 + 32 + col], sn1 = fc[s * 64 + 48 + col];
      float x0 = acc[mt][0][r] + b0;
      float x1 = acc[mt][1][r] + b1;
      float x2 = acc[mt][2][r] + b2;
      float x3 = acc[mt][3][r] + b3;
      unsigned short* op = outp + (h * S_LEN + s) * 64;
      op[col]      = f2bf((x0 * c0 - x2 * sn0) * scale);
      op[16 + col] = f2bf((x1 * c1 - x3 * sn1) * scale);
      op[32 + col] = f2bf((x2 * c0 + x0 * sn0) * scale);
      op[48 + col] = f2bf((x3 * c1 + x1 * sn1) * scale);
    }
}

// Fused V-projection + bias, stored transposed -> bf16 [HKV*64][S]
__global__ __launch_bounds__(256)
void gemm_vt_bf16_kernel(const unsigned short* __restrict__ A,
                         const unsigned short* __restrict__ Bt,
                         const float* __restrict__ bias,
                         unsigned short* __restrict__ vtOut, int K) {
  __shared__ unsigned short sB[2 * 2048];
  int wave = threadIdx.x >> 5;
  int lane = threadIdx.x & 31;
  int m0 = blockIdx.x * 256 + wave * 32;
  int n0 = blockIdx.y * 64;
  v8f acc[2][4];
  gemm_core(A, Bt, K, m0, n0, sB, acc);
  int col  = lane & 15;
  int rofs = (lane < 16) ? 0 : 8;
#pragma unroll
  for (int mt = 0; mt < 2; ++mt)
#pragma unroll
    for (int nt = 0; nt < 4; ++nt) {
      int gcol = n0 + nt * 16 + col;
      float bval = bias[gcol];
#pragma unroll
      for (int r = 0; r < 8; ++r) {
        int s = m0 + mt * 16 + r + rofs;
        vtOut[gcol * S_LEN + s] = f2bf(acc[mt][nt][r] + bval);
      }
    }
}

// ---------------------------------------------------------------------------
// Sliding-window attention with sinks. Block = (head h, query block n),
// 8 waves x 16 query rows. Scores 16x256 per wave via WMMA, band mask,
// softmax with sink, unnormalized probs -> LDS bf16 -> PV WMMA.
// ---------------------------------------------------------------------------
__global__ __launch_bounds__(256)
void attn_kernel(const unsigned short* __restrict__ qb,   // [HQ][S][64], *SCALE
                 const unsigned short* __restrict__ kb,   // [HKV][S][64]
                 const unsigned short* __restrict__ vt,   // [HKV][64][S]
                 const float* __restrict__ sinks,         // [HQ]
                 unsigned short* __restrict__ ob) {       // [S][HQ*64]
  __shared__ unsigned short sP[8][16 * 128];
  int h = blockIdx.x;
  int n = blockIdx.y;
  int hk = h >> 3;                       // kv head = h / (HQ/HKV)
  int wave = threadIdx.x >> 5;
  int lane = threadIdx.x & 31;
  int wrow = wave * 16;                  // query row base within block
  int q0   = n * W_WIN + wrow;           // global query row base
  int col  = lane & 15;
  int rofs = (lane < 16) ? 0 : 8;

  const unsigned short* qh = qb + (h * S_LEN + q0) * D_HEAD;
  const unsigned short* kh = kb + hk * S_LEN * D_HEAD;
  const unsigned short* vh = vt + hk * D_HEAD * S_LEN;

  v16bf a0 = load_fragA(qh, D_HEAD);
  v16bf a1 = load_fragA(qh + 32, D_HEAD);

  v8f sc[16];
#pragma unroll
  for (int ct = 0; ct < 16; ++ct) sc[ct] = (v8f)0.0f;

  // scores: skip tiles fully outside the causal band / before sequence start
#pragma unroll
  for (int ct = 0; ct < 16; ++ct) {
    int kpos = n * W_WIN - W_WIN + ct * 16;
    bool live = (kpos >= 0) && (ct * 16 + 15 >= wrow + 1) &&
                (ct * 16 <= wrow + 15 + W_WIN);
    if (live) {
      v16bf b0 = load_fragB(kh + kpos * D_HEAD, D_HEAD);
      v16bf b1 = load_fragB(kh + kpos * D_HEAD + 32, D_HEAD);
      sc[ct] = wmma_bf16(a0, b0, sc[ct]);
      sc[ct] = wmma_bf16(a1, b1, sc[ct]);
    }
  }

  // band mask + row max
  float rmax[8], rsum[8], inv[8];
#pragma unroll
  for (int r = 0; r < 8; ++r) rmax[r] = NEGINF;
#pragma unroll
  for (int ct = 0; ct < 16; ++ct) {
    int j = ct * 16 + col;
#pragma unroll
    for (int r = 0; r < 8; ++r) {
      int tloc = wrow + r + rofs;
      bool valid = (j >= tloc + 1) && (j <= tloc + W_WIN) && (n > 0 || j >= W_WIN);
      float v = valid ? sc[ct][r] : NEGINF;
      sc[ct][r] = v;
      rmax[r] = fmaxf(rmax[r], v);
    }
  }
  float sinkv = sinks[h];
#pragma unroll
  for (int r = 0; r < 8; ++r) {
#pragma unroll
    for (int off = 1; off < 16; off <<= 1)
      rmax[r] = fmaxf(rmax[r], __shfl_xor(rmax[r], off, 32));
    rmax[r] = fmaxf(rmax[r], sinkv);
    rsum[r] = 0.0f;
  }
#pragma unroll
  for (int ct = 0; ct < 16; ++ct)
#pragma unroll
    for (int r = 0; r < 8; ++r) {
      float e = __expf(sc[ct][r] - rmax[r]);
      sc[ct][r] = e;
      rsum[r] += e;
    }
#pragma unroll
  for (int r = 0; r < 8; ++r) {
#pragma unroll
    for (int off = 1; off < 16; off <<= 1)
      rsum[r] += __shfl_xor(rsum[r], off, 32);
    inv[r] = 1.0f / (rsum[r] + __expf(sinkv - rmax[r]));
  }

  // PV: two passes of 128 keys; per-wave LDS region (same-wave LDS is in-order)
  v8f oacc[4];
#pragma unroll
  for (int dt = 0; dt < 4; ++dt) oacc[dt] = (v8f)0.0f;
  unsigned short* myP = sP[wave];

#pragma unroll
  for (int pass = 0; pass < 2; ++pass) {
#pragma unroll
    for (int ct2 = 0; ct2 < 8; ++ct2) {
      int ct = pass * 8 + ct2;
#pragma unroll
      for (int r = 0; r < 8; ++r)
        myP[(r + rofs) * 128 + ct2 * 16 + col] = f2bf(sc[ct][r]);
    }
#pragma unroll
    for (int c2 = 0; c2 < 4; ++c2) {
      int jb = pass * 128 + c2 * 32;
      int kpos = n * W_WIN - W_WIN + jb;
      bool live = (kpos >= 0) && (jb + 31 >= wrow + 1) &&
                  (jb <= wrow + 15 + W_WIN);
      if (live) {
        v16bf pa = load_fragA(myP + c2 * 32, 128);
#pragma unroll
        for (int dt = 0; dt < 4; ++dt) {
          v16bf bv = load_fragB(vh + (dt * 16) * S_LEN + kpos, S_LEN);
          oacc[dt] = wmma_bf16(pa, bv, oacc[dt]);
        }
      }
    }
  }

#pragma unroll
  for (int dt = 0; dt < 4; ++dt)
#pragma unroll
    for (int r = 0; r < 8; ++r) {
      float o = oacc[dt][r] * inv[r];
      int srow = q0 + r + rofs;
      ob[srow * QN_C + h * D_HEAD + dt * 16 + col] = f2bf(o);
    }
}

// ---------------------------------------------------------------------------
extern "C" void kernel_launch(void* const* d_in, const int* in_sizes, int n_in,
                              void* d_out, int out_size, void* d_ws, size_t ws_size,
                              hipStream_t stream) {
  const float* x    = (const float*)d_in[0];
  const float* fc   = (const float*)d_in[1];
  const float* Wq   = (const float*)d_in[2];
  const float* bq   = (const float*)d_in[3];
  const float* Wk   = (const float*)d_in[4];
  const float* bk   = (const float*)d_in[5];
  const float* Wv   = (const float*)d_in[6];
  const float* bv   = (const float*)d_in[7];
  const float* Wo   = (const float*)d_in[8];
  const float* bo   = (const float*)d_in[9];
  const float* sinks = (const float*)d_in[10];
  float* out = (float*)d_out;
  (void)in_sizes; (void)n_in; (void)out_size; (void)ws_size;

  char* ws = (char*)d_ws;
  size_t off = 0;
  auto alloc = [&](size_t bytes) -> char* {
    char* p = ws + off;
    off += (bytes + 255) & ~(size_t)255;
    return p;
  };
  unsigned short* xb  = (unsigned short*)alloc((size_t)S_LEN * HID_C * 2);
  unsigned short* Wqt = (unsigned short*)alloc((size_t)QN_C * HID_C * 2);
  unsigned short* Wkt = (unsigned short*)alloc((size_t)KVN_C * HID_C * 2);
  unsigned short* Wvt = (unsigned short*)alloc((size_t)KVN_C * HID_C * 2);
  unsigned short* Wot = (unsigned short*)alloc((size_t)HID_C * QN_C * 2);
  unsigned short* qb  = (unsigned short*)alloc((size_t)HQ_C * S_LEN * D_HEAD * 2);
  unsigned short* kb  = (unsigned short*)alloc((size_t)HKV_C * S_LEN * D_HEAD * 2);
  unsigned short* vt  = (unsigned short*)alloc((size_t)HKV_C * D_HEAD * S_LEN * 2);
  unsigned short* ob  = (unsigned short*)alloc((size_t)S_LEN * QN_C * 2);

  // stage 0: bf16 conversions / weight transposes
  cvt_bf16_kernel<<<(S_LEN * HID_C + 255) / 256, 256, 0, stream>>>(x, xb, S_LEN * HID_C);
  cvt_transpose_kernel<<<(HID_C * QN_C + 255) / 256, 256, 0, stream>>>(Wq, Wqt, HID_C, QN_C);
  cvt_transpose_kernel<<<(HID_C * KVN_C + 255) / 256, 256, 0, stream>>>(Wk, Wkt, HID_C, KVN_C);
  cvt_transpose_kernel<<<(HID_C * KVN_C + 255) / 256, 256, 0, stream>>>(Wv, Wvt, HID_C, KVN_C);
  cvt_transpose_kernel<<<(QN_C * HID_C + 255) / 256, 256, 0, stream>>>(Wo, Wot, QN_C, HID_C);

  // stage 1: fused QKV projections (+bias, +RoPE / +transpose) in bf16 WMMA
  gemm_rope_bf16_kernel<<<dim3(S_LEN / 256, QN_C / 64), 256, 0, stream>>>(
      xb, Wqt, bq, fc, qb, HID_C, SCALE_C);
  gemm_rope_bf16_kernel<<<dim3(S_LEN / 256, KVN_C / 64), 256, 0, stream>>>(
      xb, Wkt, bk, fc, kb, HID_C, 1.0f);
  gemm_vt_bf16_kernel<<<dim3(S_LEN / 256, KVN_C / 64), 256, 0, stream>>>(
      xb, Wvt, bv, vt, HID_C);

  // stage 2: sliding-window attention with sinks (WMMA bf16)
  attn_kernel<<<dim3(HQ_C, NB_C), 256, 0, stream>>>(qb, kb, vt, sinks, ob);

  // stage 3: output projection (WMMA bf16)
  gemm_bias_f32_kernel<<<dim3(S_LEN / 256, HID_C / 64), 256, 0, stream>>>(
      ob, Wot, bo, out, HID_C, QN_C);
}
